// DomainDiscriminator_89936615178393
// MI455X (gfx1250) — matrix operand order repaired and checked by reading, not compiled, not run on hardware
//
#include <hip/hip_runtime.h>
#include <hip/hip_bf16.h>
#include <stdint.h>

typedef __attribute__((ext_vector_type(16))) _Float16 v16h;
typedef __attribute__((ext_vector_type(8)))  _Float16 v8h;
typedef __attribute__((ext_vector_type(8)))  float    v8f;

// ---------------------------------------------------------------------------
// gfx1250 async global->LDS copy (ASYNCcnt-tracked), GV addressing mode.
// LDS address taken from the vdst VGPR; per-lane 16-byte transfer.
// ---------------------------------------------------------------------------
__device__ __forceinline__ void async_ld_b128(uint32_t lds_off, const void* gaddr) {
    asm volatile("global_load_async_to_lds_b128 %0, %1, off"
                 :: "v"(lds_off), "v"((uint64_t)(uintptr_t)gaddr)
                 : "memory");
}
__device__ __forceinline__ void wait_async0() {
    asm volatile("s_wait_asynccnt 0x0" ::: "memory");
}
__device__ __forceinline__ uint32_t lds_off_of(const void* p) {
    // generic LDS address = {shared aperture, offset[31:0]}
    return (uint32_t)(uintptr_t)p;
}

// ---------------------------------------------------------------------------
// Fragment loaders from LDS tiles stored row-major [rows][32 halfwords].
// A-matrix 16x32 f16 layout (ISA 7.12.2): lanes 0-15 -> K {0..7,16..23},
// lanes 16-31 -> K {8..15,24..31}; M = lane & 15.
// ---------------------------------------------------------------------------
__device__ __forceinline__ v16h frag_ld_a(const _Float16* s) {
    int lane = threadIdx.x & 31;
    int r    = lane & 15;
    int kb   = (lane & 16) ? 8 : 0;
    const _Float16* p = s + r * 32 + kb;
    v8h lo = *(const v8h*)(p);
    v8h hi = *(const v8h*)(p + 16);
    v16h o;
#pragma unroll
    for (int i = 0; i < 8; ++i) { o[i] = lo[i]; o[i + 8] = hi[i]; }
    return o;
}

// B-matrix 32x16 (K x N) stored in LDS as [N][K]: lanes 0-15 hold K {0..7,16..23},
// lanes 16-31 hold K {8..15,24..31}; N = lane & 15.
__device__ __forceinline__ v16h frag_ld_b(const _Float16* s) {
    int lane = threadIdx.x & 31;
    int r    = lane & 15;
    int kb   = (lane & 16) ? 8 : 0;
    const _Float16* p = s + r * 32 + kb;
    v8h lo = *(const v8h*)(p);
    v8h hi = *(const v8h*)(p + 16);
    v16h o;
#pragma unroll
    for (int i = 0; i < 8; ++i) { o[i] = lo[i]; o[i + 8] = hi[i]; }
    return o;
}

// ---------------------------------------------------------------------------
// f32 -> f16 conversion (one-time streaming pass)
// ---------------------------------------------------------------------------
__global__ void f32_to_f16(const float* __restrict__ s, _Float16* __restrict__ d, int n) {
    int i = blockIdx.x * blockDim.x + threadIdx.x;
    if (i < n) d[i] = (_Float16)s[i];
}

// ---------------------------------------------------------------------------
// Implicit-GEMM conv, stride 3, pad 1, 3x3. Block = 256 thr (8 waves).
// Tile 128(M) x 128(N); K = C*9 stepped by 32. Wave grid 4(M) x 2(N).
// A: predicated f16 im2col gather, register-prefetched (no cvt in loop).
// B: f16 weights streamed via async global->LDS b128, double-buffered.
// Requires N % 128 == 0, K % 32 == 0; M is bounds-guarded.
// ---------------------------------------------------------------------------
__global__ __launch_bounds__(256)
void conv_s3_wmma(const _Float16* __restrict__ x, const _Float16* __restrict__ wh,
                  const float* __restrict__ bias, float* __restrict__ out,
                  int M, int C, int H, int W, int N, int OH, int OW) {
    __shared__ __align__(16) _Float16 As[128 * 32];
    __shared__ __align__(16) _Float16 Bs[2][128 * 32];

    const int K     = C * 9;
    const int HW    = H * W;
    const int OHW   = OH * OW;
    const int mBase = blockIdx.y * 128;
    const int nBase = blockIdx.x * 128;
    const int tid   = threadIdx.x;
    const int lane  = tid & 31;
    const int wv    = tid >> 5;
    const int wvM   = wv >> 1;   // 0..3
    const int wvN   = wv & 1;    // 0..1

    // ---- per-thread A-fill geometry (hoisted out of the K loop) ----
    const int kk  = tid & 31;    // LDS column, constant per thread
    const int lmB = tid >> 5;    // first row; rows are lmB + 8*i

    int rowBase[16];             // b*C*HW + (oh*3-1)*W + (ow*3-1)
    int rowMask[16];             // 9-bit (kh,kw) validity mask, 0 if m >= M
#pragma unroll
    for (int i = 0; i < 16; ++i) {
        int lm = lmB + 8 * i;
        int m  = mBase + lm;
        int msk = 0, base = 0;
        if (m < M) {
            int b   = m / OHW;
            int rem = m - b * OHW;
            int oh  = rem / OW;
            int ow  = rem - oh * OW;
            int oh3 = oh * 3 - 1;
            int ow3 = ow * 3 - 1;
            base = b * C * HW + oh3 * W + ow3;
#pragma unroll
            for (int kh = 0; kh < 3; ++kh)
#pragma unroll
                for (int kw = 0; kw < 3; ++kw) {
                    int ih = oh3 + kh, iw = ow3 + kw;
                    if (ih >= 0 && ih < H && iw >= 0 && iw < W)
                        msk |= 1 << (kh * 3 + kw);
                }
        }
        rowBase[i] = base;
        rowMask[i] = msk;
    }

    v8f acc[2][4];
#pragma unroll
    for (int s = 0; s < 2; ++s)
#pragma unroll
        for (int j = 0; j < 4; ++j)
#pragma unroll
            for (int i = 0; i < 8; ++i) acc[s][j][i] = 0.0f;

    _Float16 aR[16];

    // A prefetch: k0 = 0
    {
        int k   = kk;
        int c   = k / 9;
        int r9  = k - 9 * c;
        int kh  = r9 / 3;
        int kw  = r9 - 3 * kh;
        int off = c * HW + kh * W + kw;
        int bit = 1 << r9;
#pragma unroll
        for (int i = 0; i < 16; ++i)
            aR[i] = (rowMask[i] & bit) ? x[rowBase[i] + off] : (_Float16)0.0f;
    }
    // B async: k0 = 0 into buffer 0 (2 segs/thread: 128 rows x 4 b128-segs)
    {
#pragma unroll
        for (int s = 0; s < 2; ++s) {
            int seg  = tid + s * 256;
            int row  = seg >> 2;
            int segk = seg & 3;
            async_ld_b128(lds_off_of(&Bs[0][row * 32 + segk * 8]),
                          wh + (size_t)(nBase + row) * K + segk * 8);
        }
    }

    int buf = 0;
    for (int k0 = 0; k0 < K; k0 += 32) {
        // commit prefetched A slab to LDS (pure b16 stores)
#pragma unroll
        for (int i = 0; i < 16; ++i)
            As[(lmB + 8 * i) * 32 + kk] = aR[i];
        wait_async0();
        __syncthreads();

        const bool more = (k0 + 32 < K);
        if (more) {
            // next A slab global loads (hide behind DS + WMMA below)
            int k   = k0 + 32 + kk;
            int c   = k / 9;
            int r9  = k - 9 * c;
            int kh  = r9 / 3;
            int kw  = r9 - 3 * kh;
            int off = c * HW + kh * W + kw;
            int bit = 1 << r9;
#pragma unroll
            for (int i = 0; i < 16; ++i)
                aR[i] = (rowMask[i] & bit) ? x[rowBase[i] + off] : (_Float16)0.0f;
            // next B tile async into the other buffer
#pragma unroll
            for (int s = 0; s < 2; ++s) {
                int seg  = tid + s * 256;
                int row  = seg >> 2;
                int segk = seg & 3;
                async_ld_b128(lds_off_of(&Bs[buf ^ 1][row * 32 + segk * 8]),
                              wh + (size_t)(nBase + row) * K + (k0 + 32) + segk * 8);
            }
        }

        v16h a0 = frag_ld_a(As + (wvM * 32 + 0) * 32);
        v16h a1 = frag_ld_a(As + (wvM * 32 + 16) * 32);
#pragma unroll
        for (int j = 0; j < 4; ++j) {
            v16h bf = frag_ld_b(Bs[buf] + (wvN * 64 + j * 16) * 32);
            acc[0][j] = __builtin_amdgcn_wmma_f32_16x16x32_f16(
                false, a0, false, bf, (short)0, acc[0][j], false, false);
            acc[1][j] = __builtin_amdgcn_wmma_f32_16x16x32_f16(
                false, a1, false, bf, (short)0, acc[1][j], false, false);
        }
        __syncthreads();
        buf ^= 1;
    }

    // ---- epilogue: C/D layout -> VGPR i: lanes 0-15 M=i, 16-31 M=i+8 ----
    const int rowoff = (lane & 16) ? 8 : 0;
    const int ncol   = lane & 15;
    float bv[4];
    int   nOff[4];
#pragma unroll
    for (int j = 0; j < 4; ++j) {
        int n   = nBase + wvN * 64 + j * 16 + ncol;
        bv[j]   = bias[n];
        nOff[j] = n * OHW;
    }
#pragma unroll
    for (int s = 0; s < 2; ++s)
#pragma unroll
        for (int i = 0; i < 8; ++i) {
            int m = mBase + wvM * 32 + s * 16 + i + rowoff;
            if (m < M) {
                int b   = m / OHW;
                int rem = m - b * OHW;
                int t   = b * N * OHW + rem;
#pragma unroll
                for (int j = 0; j < 4; ++j)
                    out[t + nOff[j]] = acc[s][j][i] + bv[j];
            }
        }
}

// ---------------------------------------------------------------------------
// Training-mode BN stats: one block per channel; emits fused (scale, shift).
// ---------------------------------------------------------------------------
__global__ void bn_stats(const float* __restrict__ x, const float* __restrict__ g,
                         const float* __restrict__ beta, float* __restrict__ ab,
                         int C, int HW, int B) {
    const int c   = blockIdx.x;
    const int per = B * HW;
    float s = 0.0f, ss = 0.0f;
    for (int i = threadIdx.x; i < per; i += blockDim.x) {
        int b   = i / HW;
        int r   = i - b * HW;
        float v = x[(b * C + c) * HW + r];
        s += v;
        ss += v * v;
    }
    __shared__ float sm[256];
    __shared__ float sq[256];
    sm[threadIdx.x] = s;
    sq[threadIdx.x] = ss;
    __syncthreads();
    for (int o = 128; o > 0; o >>= 1) {
        if ((int)threadIdx.x < o) {
            sm[threadIdx.x] += sm[threadIdx.x + o];
            sq[threadIdx.x] += sq[threadIdx.x + o];
        }
        __syncthreads();
    }
    if (threadIdx.x == 0) {
        float inv  = 1.0f / (float)per;
        float mean = sm[0] * inv;
        float var  = sq[0] * inv - mean * mean;
        float a    = g[c] * rsqrtf(var + 1e-5f);
        ab[2 * c]     = a;
        ab[2 * c + 1] = beta[c] - mean * a;
    }
}

// normalize -> f16 copy (feeds the next stage's f16 operand stream)
__global__ void bn_apply_h(const float* __restrict__ x, const float* __restrict__ ab,
                           _Float16* __restrict__ y, int C, int HW, int total) {
    int i = blockIdx.x * blockDim.x + threadIdx.x;
    if (i >= total) return;
    int c = (i / HW) % C;
    y[i] = (_Float16)(x[i] * ab[2 * c] + ab[2 * c + 1]);
}

// ---------------------------------------------------------------------------
// M=64 GEMM, f16 in / f16 out (f32+sigmoid for the last layer).
// Block = 128 thr (4 waves), one 64x64 tile, grid.x tiles over N.
// K%32==0 layers: both tiles streamed via async global->LDS b128.
// Tail layers (K%32!=0): guarded synchronous fill.
// ---------------------------------------------------------------------------
__global__ __launch_bounds__(128)
void gemm64_wmma(const _Float16* __restrict__ Ah, const _Float16* __restrict__ Wh,
                 const float* __restrict__ bias, _Float16* __restrict__ outh,
                 float* __restrict__ outf, int N, int K, int do_sigmoid) {
    __shared__ __align__(16) _Float16 As[64 * 32];
    __shared__ __align__(16) _Float16 Bs[64 * 32];

    const int nBase   = blockIdx.x * 64;
    const int tid     = threadIdx.x;
    const int wv      = tid >> 5;
    const int lane    = tid & 31;
    const bool kalign = ((K & 31) == 0);

    if (kalign) {
        // zero the B rows that fall outside N once (asyncs never touch them)
        for (int e = tid; e < 64 * 32; e += 128) {
            int row = e >> 5;
            if (nBase + row >= N) Bs[e] = (_Float16)0.0f;
        }
    }

    v8f acc[4];
#pragma unroll
    for (int j = 0; j < 4; ++j)
#pragma unroll
        for (int i = 0; i < 8; ++i) acc[j][i] = 0.0f;

    const int kk  = tid & 31;
    const int lmB = tid >> 5;

    for (int k0 = 0; k0 < K; k0 += 32) {
        if (kalign) {
            // 64 rows x 4 b128-segs = 256 segs; 2 per thread, for A and B
#pragma unroll
            for (int s = 0; s < 2; ++s) {
                int seg  = tid + s * 128;
                int row  = seg >> 2;
                int segk = seg & 3;
                async_ld_b128(lds_off_of(&As[row * 32 + segk * 8]),
                              Ah + (size_t)row * K + k0 + segk * 8);
                int n = nBase + row;
                if (n < N)
                    async_ld_b128(lds_off_of(&Bs[row * 32 + segk * 8]),
                                  Wh + (size_t)n * K + k0 + segk * 8);
            }
            wait_async0();
        } else {
            int  k   = k0 + kk;
            bool kok = (k < K);
#pragma unroll
            for (int i = 0; i < 16; ++i) {
                int lm = lmB + 4 * i;
                As[lm * 32 + kk] = kok ? Ah[lm * K + k] : (_Float16)0.0f;
                int n = nBase + lm;
                Bs[lm * 32 + kk] =
                    (kok && n < N) ? Wh[(size_t)n * K + k] : (_Float16)0.0f;
            }
        }
        __syncthreads();

        v16h a = frag_ld_a(As + (wv << 4) * 32);
#pragma unroll
        for (int j = 0; j < 4; ++j) {
            v16h bf = frag_ld_b(Bs + (j << 4) * 32);
            acc[j]  = __builtin_amdgcn_wmma_f32_16x16x32_f16(
                false, a, false, bf, (short)0, acc[j], false, false);
        }
        __syncthreads();
    }

    const int rowoff = (lane & 16) ? 8 : 0;
    const int ncol   = lane & 15;
#pragma unroll
    for (int j = 0; j < 4; ++j) {
        int n = nBase + j * 16 + ncol;
        if (n < N) {
            float bvv = bias[n];
#pragma unroll
            for (int i = 0; i < 8; ++i) {
                int m   = wv * 16 + i + rowoff;
                float v = acc[j][i] + bvv;
                if (do_sigmoid) outf[m * N + n] = 1.0f / (1.0f + __expf(-v));
                else            outh[m * N + n] = (_Float16)v;
            }
        }
    }
}

// ---------------------------------------------------------------------------
// Host-side orchestration
// ---------------------------------------------------------------------------
extern "C" void kernel_launch(void* const* d_in, const int* in_sizes, int n_in,
                              void* d_out, int out_size, void* d_ws, size_t ws_size,
                              hipStream_t stream) {
    const float* x    = (const float*)d_in[0];
    const float* c1w  = (const float*)d_in[1];
    const float* c1b  = (const float*)d_in[2];
    const float* bn1g = (const float*)d_in[3];
    const float* bn1b = (const float*)d_in[4];
    const float* c2w  = (const float*)d_in[5];
    const float* c2b  = (const float*)d_in[6];
    const float* bn2g = (const float*)d_in[7];
    const float* bn2b = (const float*)d_in[8];

    static const int lin[12][2] = {
        {3200, 2048}, {2048, 1024}, {1024, 512}, {512, 256}, {256, 128},
        {128, 64},    {64, 32},     {32, 16},    {16, 8},    {8, 4},
        {4, 2},       {2, 1}};

    const int XELEMS  = 64 * 512 * 40 * 40;   // 52,428,800
    const int C1OUT   = 64 * 256 * 14 * 14;   //  3,211,264
    const int C2OUT   = 64 * 128 * 5 * 5;     //    204,800
    const int C1WN    = 256 * 512 * 9;        //  1,179,648
    const int C2WN    = 128 * 256 * 9;        //    294,912

    // ---- workspace layout ----
    float* ws        = (float*)d_ws;
    float* conv1_out = ws;                      // C1OUT f32
    float* ab1       = conv1_out + C1OUT;       // 512
    float* conv2_out = ab1 + 512;               // C2OUT f32
    float* ab2       = conv2_out + C2OUT;       // 256
    _Float16* harena = (_Float16*)(ab2 + 256);  // 16B-aligned

    _Float16* xh    = harena;                   // f16 input image
    _Float16* c1oh  = xh + XELEMS;              // f16 normalized conv1 out
    _Float16* a0h   = c1oh + C1OUT;             // f16 flattened activations
    _Float16* hAh   = a0h + C2OUT;              // 64*2048
    _Float16* hBh   = hAh + 64 * 2048;
    _Float16* c1wh  = hBh + 64 * 2048;
    _Float16* c2wh  = c1wh + C1WN;
    _Float16* lwh[12];
    {
        _Float16* p = c2wh + C2WN;
        for (int i = 0; i < 12; ++i) { lwh[i] = p; p += lin[i][0] * lin[i][1]; }
    }

    // ---- one-time f32 -> f16 conversions (input + all weights) ----
    f32_to_f16<<<(XELEMS + 255) / 256, 256, 0, stream>>>(x, xh, XELEMS);
    f32_to_f16<<<(C1WN + 255) / 256, 256, 0, stream>>>(c1w, c1wh, C1WN);
    f32_to_f16<<<(C2WN + 255) / 256, 256, 0, stream>>>(c2w, c2wh, C2WN);
    for (int i = 0; i < 12; ++i) {
        int n = lin[i][0] * lin[i][1];
        f32_to_f16<<<(n + 255) / 256, 256, 0, stream>>>(
            (const float*)d_in[9 + 2 * i], lwh[i], n);
    }

    // conv1: [64,512,40,40] -> [64,256,14,14]; M=12544, N=256, K=4608
    conv_s3_wmma<<<dim3(256 / 128, (12544 + 127) / 128), dim3(256), 0, stream>>>(
        xh, c1wh, c1b, conv1_out, 12544, 512, 40, 40, 256, 14, 14);
    bn_stats<<<256, 256, 0, stream>>>(conv1_out, bn1g, bn1b, ab1, 256, 196, 64);
    bn_apply_h<<<(C1OUT + 255) / 256, 256, 0, stream>>>(conv1_out, ab1, c1oh, 256, 196, C1OUT);

    // conv2: [64,256,14,14] -> [64,128,5,5]; M=1600, N=128, K=2304
    conv_s3_wmma<<<dim3(128 / 128, (1600 + 127) / 128), dim3(256), 0, stream>>>(
        c1oh, c2wh, c2b, conv2_out, 1600, 256, 14, 14, 128, 5, 5);
    bn_stats<<<128, 256, 0, stream>>>(conv2_out, bn2g, bn2b, ab2, 128, 25, 64);
    bn_apply_h<<<(C2OUT + 255) / 256, 256, 0, stream>>>(conv2_out, ab2, a0h, 128, 25, C2OUT);

    // ---- f16 MLP chain; a0h == flattened [64][3200] ----
    const _Float16* curA = a0h;
    _Float16* bufs[2]    = {hAh, hBh};
    for (int i = 0; i < 12; ++i) {
        int Kd = lin[i][0], Nd = lin[i][1];
        _Float16* oh = bufs[i & 1];
        float*    of = (float*)d_out;
        dim3 grid((Nd + 63) / 64);
        gemm64_wmma<<<grid, dim3(128), 0, stream>>>(
            curA, lwh[i], (const float*)d_in[10 + 2 * i],
            oh, of, Nd, Kd, (i == 11) ? 1 : 0);
        curA = oh;
    }
}
